// CNN_Peptide_Encoder_3805341024648
// MI455X (gfx1250) — matrix operand order repaired and checked by reading, not compiled, not run on hardware
//
#include <hip/hip_runtime.h>
#include <hip/hip_bf16.h>

typedef __attribute__((ext_vector_type(16))) _Float16 v16h;
typedef __attribute__((ext_vector_type(8)))  _Float16 v8h;
typedef __attribute__((ext_vector_type(8)))  float    v8f;

#define WPB   8              // waves per block (wave32)
#define TPB   (WPB * 32)
#define NBLK  2048           // 16384 waves -> 8 samples per wave
#define BATCH 131072
#define NEG_SLOPE 0.01f
#define LDY   24             // padded row length (48B = 16B aligned) for activation tiles

// ---- WMMA fragment builders (layouts per CDNA5 ISA 7.12.2, wave32) ----

// A-matrix 16x32 f16: lane -> M (mod 16); per lane two contiguous 8-f16 chunks
// (k0.. and k0+16.. for lane<16; +8 shifted for lane>=16) -> 2x ds_load_b128
__device__ __forceinline__ v16h frag_A(const _Float16* __restrict__ base,
                                       int lane, int m0, int k0, int ldk) {
  const int m    = m0 + (lane & 15);
  const int half = lane >> 4;
  const _Float16* row = base + m * ldk + k0;
  v16h a;
#pragma unroll
  for (int v = 0; v < 8; ++v) {
    const int kb = ((v & 4) ? 16 : 0) + (v & 3) * 2 + half * 8;
    a[2 * v]     = row[kb];
    a[2 * v + 1] = row[kb + 1];
  }
  return a;
}

// B-tile (32x16, lane -> N column, elements -> K in table order) for tap-major im2col: K = t*C + c.
// T is the (compile-time, wave-uniform) tap shift; rows are LDY-padded with zeros.
template <int T>
__device__ __forceinline__ v16h frag_B_tap(const _Float16* __restrict__ src,
                                           int lane, int c0) {
  const int nc = (lane & 15) + T;
  const int hf = lane >> 4;
  v16h b;
#pragma unroll
  for (int s = 0; s < 16; ++s) {
    const int k = (s < 8) ? (s + 8 * hf) : (8 + s + 8 * hf);
    b[s] = src[(c0 + k) * LDY + nc];
  }
  return b;
}

__device__ __forceinline__ void lds_fence() {
  asm volatile("s_wait_dscnt 0x0" ::: "memory");
}

// DPP16 butterfly step (pure VALU, no LDS): ctrl must be a literal.
#define DPP_F32(v, ctrl) \
  __int_as_float(__builtin_amdgcn_update_dpp(0, __float_as_int(v), (ctrl), 0xf, 0xf, true))

// reduce over each 16-lane row with basis masks {1,2,7,15}
__device__ __forceinline__ float row16_max(float v) {
  v = fmaxf(v, DPP_F32(v, 0xB1));   // quad_perm(1,0,3,2): xor 1
  v = fmaxf(v, DPP_F32(v, 0x4E));   // quad_perm(2,3,0,1): xor 2
  v = fmaxf(v, DPP_F32(v, 0x141));  // row_half_mirror    : xor 7
  v = fmaxf(v, DPP_F32(v, 0x140));  // row_mirror         : xor 15
  return v;
}
__device__ __forceinline__ float row16_sum(float v) {
  v += DPP_F32(v, 0xB1);
  v += DPP_F32(v, 0x4E);
  v += DPP_F32(v, 0x141);
  v += DPP_F32(v, 0x140);
  return v;
}

#define WMMA(A, B, C) __builtin_amdgcn_wmma_f32_16x16x32_f16(false, (A), false, (B), (short)0, (C), false, false)

__global__ __launch_bounds__(TPB) void pep_encoder_fused(
    const float* __restrict__ x,       // [B,21,15]
    const float* __restrict__ w0,      // [32,21]
    const float* __restrict__ attw,    // [15,32]
    const float* __restrict__ attb,    // [15]
    const float* __restrict__ w1,      // [64,32,3]
    const float* __restrict__ b1,
    const float* __restrict__ g1, const float* __restrict__ be1,
    const float* __restrict__ mu1, const float* __restrict__ va1,
    const float* __restrict__ w2,      // [20,64,3]
    const float* __restrict__ b2,
    const float* __restrict__ g2, const float* __restrict__ be2,
    const float* __restrict__ mu2, const float* __restrict__ va2,
    float* __restrict__ out)           // [B,20,11]
{
  // ---- weight staging: f16, BN scale folded, K reordered tap-major ----
  __shared__ __attribute__((aligned(16))) _Float16 sW0[32 * 32];   // [M=32][K=32] (k>=21 zero)
  __shared__ __attribute__((aligned(16))) _Float16 sW1[64 * 96];   // [M=64][K=96], K = t*32+c
  __shared__ __attribute__((aligned(16))) _Float16 sW2[32 * 192];  // [M=32][K=192], K = t*64+c, rows 20+ zero
  __shared__ float sH1[64];            // BN1 shift (bias,mean,beta folded)
  __shared__ float sH2[32];
  // ---- per-wave activation tiles (LDY-padded rows; pads stay zero) ----
  __shared__ __attribute__((aligned(16))) _Float16 sY[WPB][32 * LDY];
  __shared__ __attribute__((aligned(16))) _Float16 sZ[WPB][64 * LDY];

  const int tid = threadIdx.x;

  for (int i = tid; i < 32 * 32; i += TPB) {
    const int m = i >> 5, k = i & 31;
    sW0[i] = (k < 21) ? (_Float16)w0[m * 21 + k] : (_Float16)0.0f;
  }
  for (int i = tid; i < 64 * 96; i += TPB) {
    const int m = i / 96, r = i - 96 * m;
    const int t = r >> 5, c = r & 31;                  // K = t*32 + c
    const float inv = g1[m] * rsqrtf(va1[m] + 1e-5f);
    sW1[i] = (_Float16)(w1[m * 96 + c * 3 + t] * inv);
  }
  for (int i = tid; i < 32 * 192; i += TPB) {
    const int m = i / 192, r = i - 192 * m;
    const int t = r >> 6, c = r & 63;                  // K = t*64 + c
    float v = 0.0f;
    if (m < 20) { const float inv = g2[m] * rsqrtf(va2[m] + 1e-5f); v = w2[m * 192 + c * 3 + t] * inv; }
    sW2[i] = (_Float16)v;
  }
  for (int c = tid; c < 64; c += TPB) {
    const float inv = g1[c] * rsqrtf(va1[c] + 1e-5f);
    sH1[c] = (b1[c] - mu1[c]) * inv + be1[c];
  }
  for (int c = tid; c < 32; c += TPB) {
    float h = 0.0f;
    if (c < 20) { const float inv = g2[c] * rsqrtf(va2[c] + 1e-5f); h = (b2[c] - mu2[c]) * inv + be2[c]; }
    sH2[c] = h;
  }
  __syncthreads();

  const int wid    = tid >> 5;
  const int lane   = tid & 31;
  const int wave   = blockIdx.x * WPB + wid;
  const int nWaves = gridDim.x * WPB;
  const int half   = lane >> 4;
  const int n      = lane & 15;      // column (sequence position)

  _Float16* yb = sY[wid];
  _Float16* zb = sZ[wid];

  // zero activation tiles once; sample loop only rewrites cols 0..15
  for (int i = lane; i < 32 * LDY; i += 32) yb[i] = (_Float16)0.0f;
  for (int i = lane; i < 64 * LDY; i += 32) zb[i] = (_Float16)0.0f;

  // ---- hoist conv0 + conv1 A-fragments (112 VGPRs) ----
  const v16h a0A = frag_A(sW0, lane, 0, 0, 32);
  const v16h a0B = frag_A(sW0, lane, 16, 0, 32);
  v16h a1f[3][4];
#pragma unroll
  for (int kk = 0; kk < 3; ++kk)
#pragma unroll
    for (int mt = 0; mt < 4; ++mt)
      a1f[kk][mt] = frag_A(sW1, lane, mt * 16, kk * 32, 96);

  // ---- attention weights straight from global into registers (index clamp) ----
  const int nn = (n < 15) ? n : 14;    // lanes n==15 are masked later anyway
  float awlo[8], awhi[8];
#pragma unroll
  for (int r = 0; r < 8; ++r) {
    awlo[r] = attw[nn * 32 + r + 8 * half];
    awhi[r] = attw[nn * 32 + 16 + r + 8 * half];
  }
  const float abn = attb[nn];
  lds_fence();

  const int crow = (lane < 21) ? lane : 20;   // clamped channel for x loads (in-bounds)

  for (int b = wave; b < BATCH; b += nWaves) {
    // prefetch next sample's x while this one computes
    if (b + nWaves < BATCH)
      __builtin_prefetch(x + (size_t)(b + nWaves) * 315 + lane * 10, 0, 0);

    // ---- conv0 B-tile straight from global (lane = position, elements = input channel) ----
    const float* xs = x + (size_t)b * 315 + nn;
    v16h bx;
#pragma unroll
    for (int s = 0; s < 16; ++s) {
      const int k = (s < 8) ? (s + 8 * half) : (8 + s + 8 * half);
      bx[s] = (k < 21) ? (_Float16)xs[k * 15] : (_Float16)0.0f;
    }

    // ---- conv0: h[32x15] = W0 x X  (2 WMMA) ----
    v8f h0 = {}, h1 = {};
    h0 = WMMA(a0A, bx, h0);
    h1 = WMMA(a0B, bx, h1);

    // ---- attention: logits -> softmax over l=0..14 -> scale h ----
    float part = 0.0f;
#pragma unroll
    for (int r = 0; r < 8; ++r) part += h0[r] * awlo[r] + h1[r] * awhi[r];
    const float logit = part + __shfl_xor(part, 16, 32) + abn;   // one cross-half merge
    const float lg = (n < 15) ? logit : -3.0e38f;
    const float mx = row16_max(lg);                              // DPP butterfly, no LDS
    const float e  = (n < 15) ? __expf(lg - mx) : 0.0f;
    const float sum = row16_sum(e);                              // DPP butterfly, no LDS
    const float aw = e / sum;                                    // 0 for pad column 15

    // ---- y = h * aw, restage as f16 [C=32][LDY] ----
#pragma unroll
    for (int r = 0; r < 8; ++r) {
      const int c0 = r + 8 * half;
      yb[c0 * LDY + n]        = (_Float16)(h0[r] * aw);
      yb[(16 + c0) * LDY + n] = (_Float16)(h1[r] * aw);
    }
    lds_fence();

    // ---- conv1 (K=96 tap-major): 3 B builds (vector LDS), 12 WMMA ----
    {
      v8f acc[4] = {};
      {
        const v16h bb = frag_B_tap<0>(yb, lane, 0);
#pragma unroll
        for (int mt = 0; mt < 4; ++mt) acc[mt] = WMMA(a1f[0][mt], bb, acc[mt]);
      }
      {
        const v16h bb = frag_B_tap<1>(yb, lane, 0);
#pragma unroll
        for (int mt = 0; mt < 4; ++mt) acc[mt] = WMMA(a1f[1][mt], bb, acc[mt]);
      }
      {
        const v16h bb = frag_B_tap<2>(yb, lane, 0);
#pragma unroll
        for (int mt = 0; mt < 4; ++mt) acc[mt] = WMMA(a1f[2][mt], bb, acc[mt]);
      }
#pragma unroll
      for (int mt = 0; mt < 4; ++mt)
#pragma unroll
        for (int r = 0; r < 8; ++r) {
          const int c = mt * 16 + r + 8 * half;
          float v = acc[mt][r] + sH1[c];       // BN scale already in weights
          v = (v > 0.0f) ? v : v * NEG_SLOPE;
          zb[c * LDY + n] = (_Float16)v;
        }
    }
    lds_fence();

    // ---- conv2 (K=192 tap-major): 6 B builds, A from LDS, 12 WMMA ----
    {
      float* op = out + (size_t)b * 220;
      v8f acc[2] = {};
      // step kk: A K-offset = kk*32 ; B: tap = kk/2, channel base = (kk&1)*32
#define C2STEP(KK, T)                                                        \
      {                                                                      \
        const v16h bb = frag_B_tap<T>(zb, lane, ((KK) & 1) * 32);            \
        acc[0] = WMMA(frag_A(sW2, lane, 0,  (KK) * 32, 192), bb, acc[0]);    \
        acc[1] = WMMA(frag_A(sW2, lane, 16, (KK) * 32, 192), bb, acc[1]);    \
      }
      C2STEP(0, 0) C2STEP(1, 0) C2STEP(2, 1) C2STEP(3, 1) C2STEP(4, 2) C2STEP(5, 2)
#undef C2STEP
#pragma unroll
      for (int mt = 0; mt < 2; ++mt)
#pragma unroll
        for (int r = 0; r < 8; ++r) {
          const int c = mt * 16 + r + 8 * half;
          if (c < 20 && n < 11) {
            float v = acc[mt][r] + sH2[c];
            v = (v > 0.0f) ? v : v * NEG_SLOPE;
            op[c * 11 + n] = v;
          }
        }
    }
  }
}

extern "C" void kernel_launch(void* const* d_in, const int* in_sizes, int n_in,
                              void* d_out, int out_size, void* d_ws, size_t ws_size,
                              hipStream_t stream) {
  (void)in_sizes; (void)n_in; (void)out_size; (void)d_ws; (void)ws_size;
  const float* x    = (const float*)d_in[0];
  const float* w0   = (const float*)d_in[1];
  const float* attw = (const float*)d_in[2];
  const float* attb = (const float*)d_in[3];
  const float* w1   = (const float*)d_in[4];
  const float* b1   = (const float*)d_in[5];
  const float* g1   = (const float*)d_in[6];
  const float* be1  = (const float*)d_in[7];
  const float* mu1  = (const float*)d_in[8];
  const float* va1  = (const float*)d_in[9];
  const float* w2   = (const float*)d_in[10];
  const float* b2   = (const float*)d_in[11];
  const float* g2   = (const float*)d_in[12];
  const float* be2  = (const float*)d_in[13];
  const float* mu2  = (const float*)d_in[14];
  const float* va2  = (const float*)d_in[15];
  float* out = (float*)d_out;

  pep_encoder_fused<<<NBLK, TPB, 0, stream>>>(
      x, w0, attw, attb, w1, b1, g1, be1, mu1, va1,
      w2, b2, g2, be2, mu2, va2, out);
}
